// BNMamba_60181081752049
// MI455X (gfx1250) — matrix operand order repaired and checked by reading, not compiled, not run on hardware
//
#include <hip/hip_runtime.h>
#include <hip/hip_bf16.h>
#include <math.h>

typedef __attribute__((ext_vector_type(16))) _Float16 v16h;
typedef __attribute__((ext_vector_type(8)))  float    v8f;

#define BB    4
#define NNODE 512
#define MTOT  (BB*NNODE)   // 2048
#define EPSF  1e-5f

__device__ __forceinline__ float gelu_f(float x) {
  return 0.5f * x * (1.0f + erff(x * 0.70710678118654752f));
}
__device__ __forceinline__ float silu_f(float x) { return x / (1.0f + __expf(-x)); }
__device__ __forceinline__ float sigmoid_f(float x) { return 1.0f / (1.0f + __expf(-x)); }

__device__ __forceinline__ v16h cvt16(float4 p0, float4 p1, float4 p2, float4 p3) {
  v16h r;
  r[0]  = (_Float16)p0.x; r[1]  = (_Float16)p0.y; r[2]  = (_Float16)p0.z; r[3]  = (_Float16)p0.w;
  r[4]  = (_Float16)p1.x; r[5]  = (_Float16)p1.y; r[6]  = (_Float16)p1.z; r[7]  = (_Float16)p1.w;
  r[8]  = (_Float16)p2.x; r[9]  = (_Float16)p2.y; r[10] = (_Float16)p2.z; r[11] = (_Float16)p2.w;
  r[12] = (_Float16)p3.x; r[13] = (_Float16)p3.y; r[14] = (_Float16)p3.z; r[15] = (_Float16)p3.w;
  return r;
}

// ---------------------------------------------------------------------------
// WMMA GEMM: Y[M,N] = act(X[M,K] @ W + bias) (+ resid). One wave per 16x16
// tile; K is compile-time so the K-loop fully unrolls into K/32 WMMAs.
// W normal: W[k*ldw+n] ; TRANSW: W[n*ldw+k]. ACT: 0 none, 1 gelu, 2 silu.
// ---------------------------------------------------------------------------
template<int K, bool TRANSW, bool NGUARD, int ACT, bool BIAS, bool RESID>
__global__ void gemm_wmma_kernel(const float* __restrict__ X, int ldx,
                                 const float* __restrict__ W, int ldw,
                                 const float* __restrict__ bias,
                                 const float* __restrict__ resid, int ldr,
                                 float* __restrict__ Y, int ldy,
                                 int M, int N) {
  const int lane = threadIdx.x & 31;
  const int wave = threadIdx.x >> 5;
  const int ntn  = (N + 15) >> 4;
  const int gw   = blockIdx.x * (blockDim.x >> 5) + wave;
  const int tm   = (gw / ntn) * 16;
  const int tn   = (gw % ntn) * 16;
  if (tm >= M) return;

  const int mrow  = tm + (lane & 15);
  const int hi    = lane >> 4;
  const int ncol0 = tn + (lane & 15);
  int ncol = ncol0;
  if (NGUARD && ncol >= N) ncol = N - 1;   // clamp address only; store is guarded

  const float* xrow = X + mrow * ldx;
  v8f acc = {};
#pragma unroll
  for (int k0 = 0; k0 < K; k0 += 32) {
    // A 16x32 f16, contiguous k order used here: lane (m = lane&15) holds K run [k0+hi*16, +16)
    const float* ap = xrow + k0 + hi * 16;
    v16h a = cvt16(*(const float4*)(ap),      *(const float4*)(ap + 4),
                   *(const float4*)(ap + 8),  *(const float4*)(ap + 12));
    v16h b;
    if (TRANSW) {
      // b[kk] = W[ncol*ldw + k0 + hi*16 + kk] : 16 consecutive floats
      const float* wp = W + ncol * ldw + k0 + hi * 16;
      b = cvt16(*(const float4*)(wp),     *(const float4*)(wp + 4),
                *(const float4*)(wp + 8), *(const float4*)(wp + 12));
    } else {
      // b[kk] = W[(k0 + hi*16 + kk)*ldw + ncol] : coalesced across lanes
      const float* wp = W + (k0 + hi * 16) * ldw + ncol;
#pragma unroll
      for (int kk = 0; kk < 16; ++kk) b[kk] = (_Float16)wp[kk * ldw];
    }
    acc = __builtin_amdgcn_wmma_f32_16x16x32_f16(false, a, false, b, (short)0, acc,
                                                 false, false);
  }
  if (!NGUARD || ncol0 < N) {
    float bv = BIAS ? bias[ncol0] : 0.f;
#pragma unroll
    for (int v = 0; v < 8; ++v) {
      int row = tm + v + hi * 8;           // C/D layout: VGPR v -> row v + hi*8
      float val = acc[v] + bv;
      if (ACT == 1)      val = gelu_f(val);
      else if (ACT == 2) val = silu_f(val);
      if (RESID) val += resid[row * ldr + ncol0];
      Y[row * ldy + ncol0] = val;
    }
  }
}

// ---------------------------------------------------------------------------
// LayerNorm over last dim (=64). One wave per row, 2 elements per lane.
// ---------------------------------------------------------------------------
__global__ void ln_kernel(const float* __restrict__ X, const float* __restrict__ g,
                          const float* __restrict__ bta, float* __restrict__ Y, int M) {
  int wave = threadIdx.x >> 5;
  int lane = threadIdx.x & 31;
  int row  = blockIdx.x * (blockDim.x >> 5) + wave;
  if (row >= M) return;
  float x0 = X[row * 64 + lane];
  float x1 = X[row * 64 + 32 + lane];
  float s = x0 + x1, s2 = x0 * x0 + x1 * x1;
#pragma unroll
  for (int o = 16; o > 0; o >>= 1) { s += __shfl_xor(s, o, 32); s2 += __shfl_xor(s2, o, 32); }
  float mu  = s * (1.f / 64.f);
  float var = s2 * (1.f / 64.f) - mu * mu;
  float r   = rsqrtf(var + EPSF);
  Y[row * 64 + lane]      = (x0 - mu) * r * g[lane]      + bta[lane];
  Y[row * 64 + 32 + lane] = (x1 - mu) * r * g[lane + 32] + bta[lane + 32];
}

// deg[m] = sum_j adj[m, j]
__global__ void degree_kernel(const float* __restrict__ adj, float* __restrict__ deg) {
  int m = blockIdx.x * blockDim.x + threadIdx.x;
  if (m >= MTOT) return;
  const float* row = adj + m * NNODE;
  float s = 0.f;
  for (int j = 0; j < NNODE; ++j) s += row[j];
  deg[m] = s;
}

__global__ void embed_kernel(const float* __restrict__ deg, const float* __restrict__ We,
                             const float* __restrict__ be, float* __restrict__ x, int total) {
  int idx = blockIdx.x * blockDim.x + threadIdx.x;
  if (idx >= total) return;
  int m = idx >> 6, d = idx & 63;
  x[idx] = deg[m] * We[d] + be[d];
}

// s[b,i,d] = sum_j adj[b,i,j] * gelu(a_i[b,i,d] + a_j[b,j,d] + b1[d]); skip adj==0.
// a_i row staged into LDS via CDNA5 async global->LDS load.
__global__ void msg_accum_kernel(const float* __restrict__ adj, const float* __restrict__ ai,
                                 const float* __restrict__ aj, const float* __restrict__ b1,
                                 float* __restrict__ sbuf) {
  __shared__ float s_ai[64];
  int d = threadIdx.x;                 // 64 threads
  int m = blockIdx.x;                  // b*N + i
  int b = m / NNODE;
#if defined(__gfx1250__)
  {
    unsigned lds_off = (unsigned)(unsigned long long)(uintptr_t)(&s_ai[d]);
    unsigned long long gaddr = (unsigned long long)(uintptr_t)(ai + m * 64 + d);
    asm volatile("global_load_async_to_lds_b32 %0, %1, off"
                 :: "v"(lds_off), "v"(gaddr) : "memory");
    asm volatile("s_wait_asynccnt 0x0" ::: "memory");
  }
  __syncthreads();
  float base = s_ai[d] + b1[d];
#else
  float base = ai[m * 64 + d] + b1[d];
#endif
  const float* arow = adj + m * NNODE;
  const float* ajb  = aj + b * NNODE * 64;
  float acc = 0.f;
  for (int j = 0; j < NNODE; ++j) {
    float av = arow[j];
    if (av != 0.f) acc += av * gelu_f(base + ajb[j * 64 + d]);
  }
  sbuf[m * 64 + d] = acc;
}

// messages = (msum0 + cnt*b2) / max(cnt, 1)
__global__ void msg_fin_kernel(const float* __restrict__ msum0, const float* __restrict__ deg,
                               const float* __restrict__ b2, float* __restrict__ msg, int total) {
  int idx = blockIdx.x * blockDim.x + threadIdx.x;
  if (idx >= total) return;
  int m = idx >> 6, d = idx & 63;
  float c = deg[m];
  msg[idx] = (msum0[idx] + c * b2[d]) / fmaxf(c, 1.f);
}

// GRU-style update, in place on x
__global__ void gru_kernel(const float* __restrict__ gi, const float* __restrict__ gh,
                           float* __restrict__ x, int total) {
  int idx = blockIdx.x * blockDim.x + threadIdx.x;
  if (idx >= total) return;
  int m = idx >> 6, d = idx & 63;
  float ir = gi[m * 192 + d],       iz = gi[m * 192 + 64 + d];
  float in = gi[m * 192 + 128 + d];
  float hr = gh[m * 192 + d],       hz = gh[m * 192 + 64 + d];
  float hn = gh[m * 192 + 128 + d];
  float r = sigmoid_f(ir + hr), z = sigmoid_f(iz + hz);
  float n = tanhf(in + r * hn);
  float xv = x[idx];
  x[idx] = xv + (1.f - z) * n + z * xv;
}

// cat = [x, sys_emb[fs]]
__global__ void concat_kernel(const float* __restrict__ x, const float* __restrict__ sys_emb,
                              const int* __restrict__ fs, float* __restrict__ cat, int total) {
  int idx = blockIdx.x * blockDim.x + threadIdx.x;
  if (idx >= total) return;
  int m = idx >> 6, d = idx & 63;
  cat[m * 128 + d] = x[idx];
  int v = fs[m] & 7;
  cat[m * 128 + 64 + d] = sys_emb[v * 64 + d];
}

// stable counting sort per batch (values 0..6)
__global__ void sort_kernel(const int* __restrict__ fs, int* __restrict__ sidx) {
  int b = threadIdx.x;
  if (b >= BB) return;
  int cnt[8];
#pragma unroll
  for (int v = 0; v < 8; ++v) cnt[v] = 0;
  for (int i = 0; i < NNODE; ++i) cnt[fs[b * NNODE + i] & 7]++;
  int off[8]; int acc = 0;
#pragma unroll
  for (int v = 0; v < 8; ++v) { off[v] = acc; acc += cnt[v]; }
  for (int i = 0; i < NNODE; ++i) {
    int v = fs[b * NNODE + i] & 7;
    sidx[b * NNODE + off[v]++] = i;
  }
}

__global__ void gather_kernel(const float* __restrict__ src, const int* __restrict__ sidx,
                              float* __restrict__ dst, int total) {
  int idx = blockIdx.x * blockDim.x + threadIdx.x;
  if (idx >= total) return;
  int m = idx >> 6, d = idx & 63;
  int b = m / NNODE;
  int i = sidx[m];
  dst[idx] = src[(b * NNODE + i) * 64 + d];
}

// causal depthwise conv (K=4) over first 128 channels of xz, + silu
__global__ void conv_silu_kernel(const float* __restrict__ xz, const float* __restrict__ cw,
                                 const float* __restrict__ cb, float* __restrict__ out, int total) {
  int idx = blockIdx.x * blockDim.x + threadIdx.x;
  if (idx >= total) return;
  int c = idx & 127, m = idx >> 7;
  int t = m % NNODE, b = m / NNODE;
  float acc = cb[c];
#pragma unroll
  for (int k = 0; k < 4; ++k) {
    int tt = t - 3 + k;
    if (tt >= 0) acc += xz[(b * NNODE + tt) * 256 + c] * cw[c * 4 + k];
  }
  out[idx] = silu_f(acc);
}

// dt = softplus(xproj[..., :4] @ dt_W + dt_b)
__global__ void dt_kernel(const float* __restrict__ xproj, const float* __restrict__ dtW,
                          const float* __restrict__ dtb, float* __restrict__ dt, int total) {
  int idx = blockIdx.x * blockDim.x + threadIdx.x;
  if (idx >= total) return;
  int c = idx & 127, m = idx >> 7;
  float acc = dtb[c];
#pragma unroll
  for (int r = 0; r < 4; ++r) acc += xproj[m * 132 + r] * dtW[r * 128 + c];
  dt[idx] = (acc > 20.f) ? acc : log1pf(__expf(acc));
}

// selective scan: 1 block / batch, 512 threads, 4 lanes per channel x 16 state
// elems. Per-step B/C vectors staged once into LDS instead of 128x-redundant
// global reads.
__global__ void scan_kernel(const float* __restrict__ xproj, const float* __restrict__ dt,
                            const float* __restrict__ u_, const float* __restrict__ xz,
                            const float* __restrict__ A_log, const float* __restrict__ Dp,
                            float* __restrict__ yv) {
  __shared__ float sBC[128];           // [0:64) = B_t, [64:128) = C_t
  int b = blockIdx.x;
  int t = threadIdx.x;
  int d = t >> 2;
  int q = t & 3;
  int s0 = q * 16;
  float A[16], h[16];
#pragma unroll
  for (int j = 0; j < 16; ++j) { A[j] = -__expf(A_log[d * 64 + s0 + j]); h[j] = 0.f; }
  float Dv = Dp[d];
  for (int step = 0; step < NNODE; ++step) {
    int m = b * NNODE + step;
    if (t < 128) sBC[t] = xproj[m * 132 + 4 + t];
    __syncthreads();
    float dtv = dt[m * 128 + d];
    float uu  = u_[m * 128 + d];
    float du  = dtv * uu;
    float y = 0.f;
#pragma unroll
    for (int j = 0; j < 16; ++j) {
      float dA = __expf(dtv * A[j]);
      h[j] = dA * h[j] + du * sBC[s0 + j];
      y += h[j] * sBC[64 + s0 + j];
    }
    y += __shfl_xor(y, 1, 32);
    y += __shfl_xor(y, 2, 32);
    if (q == 0) {
      float zz = xz[m * 256 + 128 + d];
      yv[m * 128 + d] = (y + uu * Dv) * silu_f(zz);
    }
    __syncthreads();
  }
}

// mean over nodes (perm-invariant, argsort-inverse skipped) + LN + gelu(g@W_r + b_r)
__global__ void readout_kernel(const float* __restrict__ xs, const float* __restrict__ lng,
                               const float* __restrict__ lnb, const float* __restrict__ Wr,
                               const float* __restrict__ br, float* __restrict__ out) {
  __shared__ float gbuf[BB * 64];
  __shared__ float gln[BB * 64];
  int tid = threadIdx.x;               // 256
  int b = tid >> 6, d = tid & 63;
  float acc = 0.f;
  for (int i = 0; i < NNODE; ++i) acc += xs[(b * NNODE + i) * 64 + d];
  gbuf[tid] = acc * (1.f / NNODE);
  __syncthreads();
  float s = 0.f, s2 = 0.f;
  for (int j = 0; j < 64; ++j) { float v = gbuf[b * 64 + j]; s += v; s2 += v * v; }
  float mu = s / 64.f, var = s2 / 64.f - mu * mu;
  float r = rsqrtf(var + EPSF);
  gln[tid] = (gbuf[tid] - mu) * r * lng[d] + lnb[d];
  __syncthreads();
  float o = br[d];
  for (int j = 0; j < 64; ++j) o += gln[b * 64 + j] * Wr[j * 64 + d];
  out[tid] = gelu_f(o);
}

// ---------------------------------------------------------------------------
#define GEMM_LAUNCH(KV, TW, NG, ACTV, BIASV, RESV, X_, ldx_, W_, ldw_, bias_, res_, ldr_, Y_, ldy_, Ni) \
  do {                                                                                    \
    int ntn_ = ((Ni) + 15) / 16;                                                          \
    int blocks_ = ((MTOT / 16) * ntn_ + 3) / 4;                                           \
    gemm_wmma_kernel<KV, TW, NG, ACTV, BIASV, RESV><<<blocks_, 128, 0, stream>>>(         \
        X_, ldx_, W_, ldw_, bias_, res_, ldr_, Y_, ldy_, MTOT, Ni);                       \
  } while (0)

extern "C" void kernel_launch(void* const* d_in, const int* in_sizes, int n_in,
                              void* d_out, int out_size, void* d_ws, size_t ws_size,
                              hipStream_t stream) {
  (void)in_sizes; (void)n_in; (void)out_size; (void)ws_size;
  // pytree flatten order (sorted dict keys, lists in order):
  const float* adj     = (const float*)d_in[0];
  const int*   fs      = (const int*)d_in[1];
  const float* W_emb   = (const float*)d_in[2];
  const float* W_r     = (const float*)d_in[3];
  const float* Wp      = (const float*)d_in[4];
  const float* b_emb   = (const float*)d_in[5];
  const float* b_r     = (const float*)d_in[6];
  const float* bp      = (const float*)d_in[7];
  const float* ro_ln_b = (const float*)d_in[28];
  const float* ro_ln_g = (const float*)d_in[29];
  const float* sys_emb = (const float*)d_in[56];

  const int M = MTOT;
  float* w = (float*)d_ws;
  size_t off = 0;
  float* deg   = w + off; off += M;
  float* x     = w + off; off += (size_t)M * 64;
  float* xn    = w + off; off += (size_t)M * 64;
  float* a_i   = w + off; off += (size_t)M * 64;
  float* a_j   = w + off; off += (size_t)M * 64;
  float* sbuf  = w + off; off += (size_t)M * 64;
  float* msum0 = w + off; off += (size_t)M * 64;
  float* msg   = w + off; off += (size_t)M * 64;
  float* gi    = w + off; off += (size_t)M * 192;
  float* gh    = w + off; off += (size_t)M * 192;
  float* cat   = w + off; off += (size_t)M * 128;
  float* xs_a  = w + off; off += (size_t)M * 64;
  float* xs_b  = w + off; off += (size_t)M * 64;
  int*   sidx  = (int*)(w + off); off += M;
  float* xz    = w + off; off += (size_t)M * 256;
  float* xconv = w + off; off += (size_t)M * 128;
  float* xproj = w + off; off += (size_t)M * 132;
  float* dtb   = w + off; off += (size_t)M * 128;
  float* yv    = w + off; off += (size_t)M * 128;

  auto ew = [](int n) { return (n + 255) / 256; };

  // ---- embedding ----
  degree_kernel<<<(M + 255) / 256, 256, 0, stream>>>(adj, deg);
  embed_kernel<<<ew(M * 64), 256, 0, stream>>>(deg, W_emb, b_emb, x, M * 64);

  // ---- MPNN layers ----
  for (int l = 0; l < 2; ++l) {
    int base = 8 + 10 * l;
    const float* W1   = (const float*)d_in[base + 0];
    const float* W2   = (const float*)d_in[base + 1];
    const float* W_hh = (const float*)d_in[base + 2];
    const float* W_ih = (const float*)d_in[base + 3];
    const float* b1   = (const float*)d_in[base + 4];
    const float* b2   = (const float*)d_in[base + 5];
    const float* b_hh = (const float*)d_in[base + 6];
    const float* b_ih = (const float*)d_in[base + 7];
    const float* ln_b = (const float*)d_in[base + 8];
    const float* ln_g = (const float*)d_in[base + 9];

    ln_kernel<<<(M + 3) / 4, 128, 0, stream>>>(x, ln_g, ln_b, xn, M);
    GEMM_LAUNCH(64, false, false, 0, false, false, xn, 64, W1,           64, nullptr, nullptr, 0, a_i, 64, 64);
    GEMM_LAUNCH(64, false, false, 0, false, false, xn, 64, W1 + 64 * 64, 64, nullptr, nullptr, 0, a_j, 64, 64);
    msg_accum_kernel<<<M, 64, 0, stream>>>(adj, a_i, a_j, b1, sbuf);
    GEMM_LAUNCH(64, false, false, 0, false, false, sbuf, 64, W2, 64, nullptr, nullptr, 0, msum0, 64, 64);
    msg_fin_kernel<<<ew(M * 64), 256, 0, stream>>>(msum0, deg, b2, msg, M * 64);
    GEMM_LAUNCH(64, true, false, 0, true, false, msg, 64, W_ih, 64, b_ih, nullptr, 0, gi, 192, 192);
    GEMM_LAUNCH(64, true, false, 0, true, false, x,   64, W_hh, 64, b_hh, nullptr, 0, gh, 192, 192);
    gru_kernel<<<ew(M * 64), 256, 0, stream>>>(gi, gh, x, M * 64);
  }

  // ---- projection + sort ----
  concat_kernel<<<ew(M * 64), 256, 0, stream>>>(x, sys_emb, fs, cat, M * 64);
  GEMM_LAUNCH(128, false, false, 1 /*gelu*/, true, false, cat, 128, Wp, 64, bp, nullptr, 0, xs_a, 64, 64);
  sort_kernel<<<1, 32, 0, stream>>>(fs, sidx);
  gather_kernel<<<ew(M * 64), 256, 0, stream>>>(xs_a, sidx, xs_b, M * 64);

  float* xs_cur = xs_b;
  float* xs_nxt = xs_a;

  // ---- SSM blocks ----
  for (int l = 0; l < 2; ++l) {
    int base = 30 + 13 * l;
    const float* A_log  = (const float*)d_in[base + 0];
    const float* Dp     = (const float*)d_in[base + 1];
    const float* W_in   = (const float*)d_in[base + 2];
    const float* W_out  = (const float*)d_in[base + 3];
    const float* W_x    = (const float*)d_in[base + 4];
    const float* b_in   = (const float*)d_in[base + 5];
    const float* b_out  = (const float*)d_in[base + 6];
    const float* conv_b = (const float*)d_in[base + 7];
    const float* conv_w = (const float*)d_in[base + 8];
    const float* dt_W   = (const float*)d_in[base + 9];
    const float* dt_b   = (const float*)d_in[base + 10];
    const float* ln_b   = (const float*)d_in[base + 11];
    const float* ln_g   = (const float*)d_in[base + 12];

    ln_kernel<<<(M + 3) / 4, 128, 0, stream>>>(xs_cur, ln_g, ln_b, xn, M);
    GEMM_LAUNCH(64, false, false, 0, true, false, xn, 64, W_in, 256, b_in, nullptr, 0, xz, 256, 256);
    conv_silu_kernel<<<ew(M * 128), 256, 0, stream>>>(xz, conv_w, conv_b, xconv, M * 128);
    GEMM_LAUNCH(128, false, true, 0, false, false, xconv, 128, W_x, 132, nullptr, nullptr, 0, xproj, 132, 132);
    dt_kernel<<<ew(M * 128), 256, 0, stream>>>(xproj, dt_W, dt_b, dtb, M * 128);
    scan_kernel<<<BB, 512, 0, stream>>>(xproj, dtb, xconv, xz, A_log, Dp, yv);
    GEMM_LAUNCH(128, false, false, 0, true, true, yv, 128, W_out, 64, b_out, xs_cur, 64, xs_nxt, 64, 64);
    float* tmp = xs_cur; xs_cur = xs_nxt; xs_nxt = tmp;
  }

  // ---- readout ----
  readout_kernel<<<1, 256, 0, stream>>>(xs_cur, ro_ln_g, ro_ln_b, W_r, b_r, (float*)d_out);
}
